// BaselineModel_43250320670958
// MI455X (gfx1250) — compile-verified, matrix operand order checked by reading, not run on hardware
//
#include <hip/hip_runtime.h>
#include <math.h>

// Problem constants (from the reference)
constexpr int L_  = 1024;
constexpr int D_  = 768;
constexpr int DI_ = 1536;
constexpr int N_  = 16;
constexpr int R_  = 48;
constexpr int V_  = 50280;
constexpr int NL_ = 24;
constexpr int XD_ = R_ + 2 * N_;   // 80, x_dbl row width

typedef __attribute__((ext_vector_type(16))) __bf16 v16bf;
typedef __attribute__((ext_vector_type(8)))  float  v8f;

// ---------------------------------------------------------------------------
// Embedding gather: x[l,:] = emb[ids[l],:], res = 0
// ---------------------------------------------------------------------------
__global__ void k_embed(const int* __restrict__ ids, const float* __restrict__ emb,
                        float* __restrict__ x, float* __restrict__ res) {
  int row = blockIdx.x;
  int tok = ids[row];
  for (int i = threadIdx.x; i < D_; i += 256) {
    x[(size_t)row * D_ + i]   = emb[(size_t)tok * D_ + i];
    res[(size_t)row * D_ + i] = 0.0f;
  }
}

// ---------------------------------------------------------------------------
// res += x ; h = rmsnorm(res) * nw        (one block per row, D_=768)
// ---------------------------------------------------------------------------
__global__ void k_addnorm(const float* __restrict__ x, float* __restrict__ res,
                          const float* __restrict__ nw, float* __restrict__ h) {
  __shared__ float sred[256];
  int row = blockIdx.x;
  float local[3];
  float ss = 0.0f;
#pragma unroll
  for (int j = 0; j < 3; ++j) {
    int i = threadIdx.x + j * 256;
    float v = x[(size_t)row * D_ + i] + res[(size_t)row * D_ + i];
    res[(size_t)row * D_ + i] = v;
    local[j] = v;
    ss += v * v;
  }
  sred[threadIdx.x] = ss;
  __syncthreads();
  for (int s = 128; s > 0; s >>= 1) {
    if (threadIdx.x < s) sred[threadIdx.x] += sred[threadIdx.x + s];
    __syncthreads();
  }
  float scale = rsqrtf(sred[0] / (float)D_ + 1e-5f);
#pragma unroll
  for (int j = 0; j < 3; ++j) {
    int i = threadIdx.x + j * 256;
    h[(size_t)row * D_ + i] = local[j] * scale * nw[i];
  }
}

// ---------------------------------------------------------------------------
// Fragment loader: lane's 16 bf16 halves are two contiguous 8-float runs at
// row[k0+kg*8] and row[k0+16+kg*8]  ->  4 x b128 loads, no per-element guards.
// Guards fold to constants after the (compile-time-K) k-loop is unrolled.
// ---------------------------------------------------------------------------
template<int K>
__device__ __forceinline__ v16bf load_frag(const float* __restrict__ row,
                                           int k0, int kg) {
  int b0 = k0 + kg * 8;
  int b1 = k0 + 16 + kg * 8;
  float4 x0, x1, x2, x3;
  if ((K % 32 == 0) || (k0 + 16 <= K)) {
    x0 = *(const float4*)(row + b0);
    x1 = *(const float4*)(row + b0 + 4);
  } else {
    x0 = make_float4(0.f, 0.f, 0.f, 0.f);
    x1 = x0;
  }
  if ((K % 32 == 0) || (k0 + 32 <= K)) {
    x2 = *(const float4*)(row + b1);
    x3 = *(const float4*)(row + b1 + 4);
  } else {
    x2 = make_float4(0.f, 0.f, 0.f, 0.f);
    x3 = x2;
  }
  v16bf f;
  f[0]  = (__bf16)x0.x; f[1]  = (__bf16)x0.y; f[2]  = (__bf16)x0.z; f[3]  = (__bf16)x0.w;
  f[4]  = (__bf16)x1.x; f[5]  = (__bf16)x1.y; f[6]  = (__bf16)x1.z; f[7]  = (__bf16)x1.w;
  f[8]  = (__bf16)x2.x; f[9]  = (__bf16)x2.y; f[10] = (__bf16)x2.z; f[11] = (__bf16)x2.w;
  f[12] = (__bf16)x3.x; f[13] = (__bf16)x3.y; f[14] = (__bf16)x3.z; f[15] = (__bf16)x3.w;
  return f;
}

// ---------------------------------------------------------------------------
// WMMA GEMM: C[M, NT*16*gridX] = A[M,K] * W[N,K]^T (fp32 in, bf16 WMMA, f32 acc)
// One wave computes a 16 x (16*NT) strip: 1 A fragment feeds NT B fragments.
// EPI==1: C = softplus(C + bias[col]).
// ---------------------------------------------------------------------------
template<int K, int NT, int EPI>
__global__ void k_gemm(const float* __restrict__ A, const float* __restrict__ W,
                       const float* __restrict__ bias, float* __restrict__ C,
                       int lda, int ldw, int ldc) {
  int lane   = threadIdx.x & 31;   // wave32
  int mTile  = blockIdx.y * 16;
  int nTile0 = blockIdx.x * (16 * NT);
  int ml = lane & 15;
  int kg = lane >> 4;

  const float* aRow = A + (size_t)(mTile + ml) * lda;
  const float* wRow[NT];
#pragma unroll
  for (int t = 0; t < NT; ++t)
    wRow[t] = W + (size_t)(nTile0 + t * 16 + ml) * ldw;

  v8f acc[NT];
#pragma unroll
  for (int t = 0; t < NT; ++t) acc[t] = (v8f){};

#pragma unroll 2
  for (int k0 = 0; k0 < K; k0 += 32) {
    if (k0 + 32 < K) __builtin_prefetch(aRow + k0 + 32, 0, 3);
    v16bf af = load_frag<K>(aRow, k0, kg);
#pragma unroll
    for (int t = 0; t < NT; ++t) {
      v16bf bf = load_frag<K>(wRow[t], k0, kg);
      acc[t] = __builtin_amdgcn_wmma_f32_16x16x32_bf16(
          false, af, false, bf, (short)0, acc[t], false, false);
    }
  }

#pragma unroll
  for (int t = 0; t < NT; ++t) {
    int col = nTile0 + t * 16 + ml;
    float b = (EPI == 1) ? bias[col] : 0.0f;
#pragma unroll
    for (int r = 0; r < 8; ++r) {
      int row = mTile + r + kg * 8;  // C/D layout: VGPR r, lanes>=16 hold M=r+8
      float v = acc[t][r];
      if (EPI == 1) {
        v += b;
        v = (v > 20.0f) ? v : log1pf(expf(v));   // softplus
      }
      C[(size_t)row * ldc + col] = v;
    }
  }
}

// ---------------------------------------------------------------------------
// Depthwise causal conv (K=4) + bias + SiLU; input is x-half of xz (stride 2*DI)
// ---------------------------------------------------------------------------
__global__ void k_conv(const float* __restrict__ xz, const float* __restrict__ cw,
                       const float* __restrict__ cb, float* __restrict__ xs) {
  int idx = blockIdx.x * 256 + threadIdx.x;
  if (idx >= L_ * DI_) return;
  int l = idx / DI_;
  int d = idx - l * DI_;
  float v = 0.0f;
#pragma unroll
  for (int k = 0; k < 4; ++k) {
    int ls = l - 3 + k;
    if (ls >= 0) v += cw[d * 4 + k] * xz[(size_t)ls * (2 * DI_) + d];
  }
  v += cb[d];
  xs[idx] = v / (1.0f + expf(-v));   // silu
}

// ---------------------------------------------------------------------------
// Selective scan: one thread per channel d, 16 states in registers, serial in L.
// y = (scan + x*Dp) * silu(z)
// ---------------------------------------------------------------------------
__global__ void k_scan(const float* __restrict__ dt, const float* __restrict__ xs,
                       const float* __restrict__ xz, const float* __restrict__ xdbl,
                       const float* __restrict__ Al, const float* __restrict__ Dp,
                       float* __restrict__ y) {
  int d = blockIdx.x * 256 + threadIdx.x;
  if (d >= DI_) return;
  float A[N_], h[N_];
#pragma unroll
  for (int n = 0; n < N_; ++n) {
    A[n] = -expf(Al[(size_t)d * N_ + n]);
    h[n] = 0.0f;
  }
  float Dd = Dp[d];
  for (int l = 0; l < L_; ++l) {
    float dtl = dt[(size_t)l * DI_ + d];
    float xl  = xs[(size_t)l * DI_ + d];
    float zl  = xz[(size_t)l * (2 * DI_) + DI_ + d];
    const float* bc = xdbl + (size_t)l * XD_ + R_;   // [B(16) | C(16)]
    float dx  = dtl * xl;
    float acc = 0.0f;
#pragma unroll
    for (int n = 0; n < N_; ++n) {
      float hn = expf(dtl * A[n]) * h[n] + dx * bc[n];
      h[n] = hn;
      acc += hn * bc[N_ + n];
    }
    float v = acc + xl * Dd;
    y[(size_t)l * DI_ + d] = v * (zl / (1.0f + expf(-zl)));
  }
}

// ---------------------------------------------------------------------------
// Final rmsnorm of (x+res) last row only
// ---------------------------------------------------------------------------
__global__ void k_final(const float* __restrict__ x, const float* __restrict__ res,
                        const float* __restrict__ nfw, float* __restrict__ xf) {
  __shared__ float sred[256];
  const int row = L_ - 1;
  float local[3];
  float ss = 0.0f;
#pragma unroll
  for (int j = 0; j < 3; ++j) {
    int i = threadIdx.x + j * 256;
    float v = x[(size_t)row * D_ + i] + res[(size_t)row * D_ + i];
    local[j] = v;
    ss += v * v;
  }
  sred[threadIdx.x] = ss;
  __syncthreads();
  for (int s = 128; s > 0; s >>= 1) {
    if (threadIdx.x < s) sred[threadIdx.x] += sred[threadIdx.x + s];
    __syncthreads();
  }
  float scale = rsqrtf(sred[0] / (float)D_ + 1e-5f);
#pragma unroll
  for (int j = 0; j < 3; ++j) {
    int i = threadIdx.x + j * 256;
    xf[i] = local[j] * scale * nfw[i];
  }
}

__global__ void k_fill(float* __restrict__ out, int n) {
  int i = blockIdx.x * 256 + threadIdx.x;
  if (i < n) out[i] = -__builtin_inff();
}

// Only the <=1024 tokens present in the input get finite logits (the mask
// kills everything else), so compute just those dot products.
__global__ void k_logits(const int* __restrict__ ids, const float* __restrict__ xf,
                         const float* __restrict__ lm, float* __restrict__ out) {
  __shared__ float sred[256];
  int l = blockIdx.x;
  int v = ids[l];
  float s = 0.0f;
  for (int i = threadIdx.x; i < D_; i += 256)
    s += xf[i] * lm[(size_t)v * D_ + i];
  sred[threadIdx.x] = s;
  __syncthreads();
  for (int st = 128; st > 0; st >>= 1) {
    if (threadIdx.x < st) sred[threadIdx.x] += sred[threadIdx.x + st];
    __syncthreads();
  }
  if (threadIdx.x == 0) out[v] = sred[0];   // duplicates write identical values
}

// ---------------------------------------------------------------------------
extern "C" void kernel_launch(void* const* d_in, const int* in_sizes, int n_in,
                              void* d_out, int out_size, void* d_ws, size_t ws_size,
                              hipStream_t stream) {
  (void)in_sizes; (void)n_in; (void)out_size; (void)ws_size;
  const int*   ids = (const int*)d_in[0];
  const float* emb = (const float*)d_in[1];
  const float* inw = (const float*)d_in[2];   // (NL, 2*DI, D)
  const float* cw  = (const float*)d_in[3];   // (NL, DI, 4)
  const float* cb  = (const float*)d_in[4];   // (NL, DI)
  const float* xpw = (const float*)d_in[5];   // (NL, 80, DI)
  const float* dtw = (const float*)d_in[6];   // (NL, DI, R)
  const float* dtb = (const float*)d_in[7];   // (NL, DI)
  const float* Al  = (const float*)d_in[8];   // (NL, DI, N)
  const float* Dp  = (const float*)d_in[9];   // (NL, DI)
  const float* ow  = (const float*)d_in[10];  // (NL, D, DI)
  const float* nw  = (const float*)d_in[11];  // (NL, D)
  const float* nfw = (const float*)d_in[12];  // (D)
  const float* lmw = (const float*)d_in[13];  // (V, D)
  float* out = (float*)d_out;

  // Workspace carve-up (~41 MB of fp32)
  float* ws   = (float*)d_ws;
  float* x    = ws; ws += (size_t)L_ * D_;
  float* res  = ws; ws += (size_t)L_ * D_;
  float* h    = ws; ws += (size_t)L_ * D_;
  float* xz   = ws; ws += (size_t)L_ * 2 * DI_;
  float* xs   = ws; ws += (size_t)L_ * DI_;
  float* xdbl = ws; ws += (size_t)L_ * XD_;
  float* dt   = ws; ws += (size_t)L_ * DI_;
  float* y    = ws; ws += (size_t)L_ * DI_;
  float* xf   = ws; ws += D_;

  k_embed<<<L_, 256, 0, stream>>>(ids, emb, x, res);

  for (int lyr = 0; lyr < NL_; ++lyr) {
    const float* inw_l = inw + (size_t)lyr * 2 * DI_ * D_;
    const float* cw_l  = cw  + (size_t)lyr * DI_ * 4;
    const float* cb_l  = cb  + (size_t)lyr * DI_;
    const float* xpw_l = xpw + (size_t)lyr * XD_ * DI_;
    const float* dtw_l = dtw + (size_t)lyr * DI_ * R_;
    const float* dtb_l = dtb + (size_t)lyr * DI_;
    const float* Al_l  = Al  + (size_t)lyr * DI_ * N_;
    const float* Dp_l  = Dp  + (size_t)lyr * DI_;
    const float* ow_l  = ow  + (size_t)lyr * D_ * DI_;
    const float* nw_l  = nw  + (size_t)lyr * D_;

    // res += x ; h = rmsnorm(res)*nw
    k_addnorm<<<L_, 256, 0, stream>>>(x, res, nw_l, h);

    // xz = h @ in_w^T        (1024 x 768 x 3072)  K=768, NT=4
    k_gemm<768, 4, 0><<<dim3(2 * DI_ / 64, L_ / 16), 32, 0, stream>>>(
        h, inw_l, nullptr, xz, D_, D_, 2 * DI_);

    // xs = silu(conv(x) + cb)
    k_conv<<<(L_ * DI_) / 256, 256, 0, stream>>>(xz, cw_l, cb_l, xs);

    // x_dbl = xs @ xp_w^T    (1024 x 1536 x 80)   K=1536, NT=5 (whole 80 cols)
    k_gemm<1536, 5, 0><<<dim3(1, L_ / 16), 32, 0, stream>>>(
        xs, xpw_l, nullptr, xdbl, DI_, DI_, XD_);

    // dt = softplus(x_dbl[:, :R] @ dtw^T + dtb)   K=48 (zero-padded), NT=4
    k_gemm<48, 4, 1><<<dim3(DI_ / 64, L_ / 16), 32, 0, stream>>>(
        xdbl, dtw_l, dtb_l, dt, XD_, R_, DI_);

    // selective scan + skip + silu(z) gate -> y
    k_scan<<<DI_ / 256, 256, 0, stream>>>(dt, xs, xz, xdbl, Al_l, Dp_l, y);

    // x = y @ ow^T           (1024 x 1536 x 768)  K=1536, NT=4
    k_gemm<1536, 4, 0><<<dim3(D_ / 64, L_ / 16), 32, 0, stream>>>(
        y, ow_l, nullptr, x, DI_, DI_, D_);
  }

  k_final<<<1, 256, 0, stream>>>(x, res, nfw, xf);
  k_fill<<<(V_ + 255) / 256, 256, 0, stream>>>(out, V_);
  k_logits<<<L_, 256, 0, stream>>>(ids, xf, lmw, out);
}
